// NONLocal_Correlation_37014028157090
// MI455X (gfx1250) — matrix operand order repaired and checked by reading, not compiled, not run on hardware
//
#include <hip/hip_runtime.h>
#include <hip/hip_bf16.h>
#include <math.h>

// ---------------------------------------------------------------------------
// Non-local (attention) block for MI455X / gfx1250, wave32 + WMMA bf16.
//   B=2, C=256, N=8192, CI=256
// All four GEMMs (3 projections, attention S & O, W-conv) run on
// v_wmma_f32_16x16x32_bf16. Flash-attention core uses CDNA5 async
// global->LDS loads (ASYNCcnt) with a double-buffered K tile.
// Whole working set (<60MB) is L2-resident (192MB).
// ---------------------------------------------------------------------------

typedef __bf16 bf16;
typedef bf16  bf16x8  __attribute__((ext_vector_type(8)));
typedef bf16  bf16x16 __attribute__((ext_vector_type(16)));
typedef float floatx8 __attribute__((ext_vector_type(8)));
typedef int   v4i     __attribute__((vector_size(16)));

#define B_   2
#define C_   256
#define N_   8192
#define CI_  256
#define EPS_ 1e-5f

// tiling
#define BM  64        // rows per workgroup tile
#define BN  64        // key tile
#define D_  256       // head dim (= CI)
#define QLD 264       // 64x256 bf16 tile row stride (+8 pad)
#define KLD 264
#define GLD 72        // G tile stored [c][m]: 256 x 64 (+8 pad)
#define SLD 68        // S score tile fp32 row stride (+4 pad)
#define PLD 136       // P bf16 tile ALIASED over S rows (136*2B == 68*4B)

// ---------------------------------------------------------------------------
// CDNA5 async global->LDS path (guarded; falls back to sync copies)
// ---------------------------------------------------------------------------
#if defined(__has_builtin)
# if __has_builtin(__builtin_amdgcn_global_load_async_to_lds_b128)
#  define HAVE_ASYNC 1
# endif
#endif
#ifndef HAVE_ASYNC
# define HAVE_ASYNC 0
#endif

#define AS1 __attribute__((address_space(1)))
#define AS3 __attribute__((address_space(3)))

__device__ __forceinline__ void cp16_async(void* ldst, const void* gsrc) {
#if HAVE_ASYNC
    // emits global_load_async_to_lds_b128 (tracked by ASYNCcnt)
    __builtin_amdgcn_global_load_async_to_lds_b128(
        (AS1 v4i*)(uintptr_t)gsrc,
        (AS3 v4i*)(uint32_t)(uintptr_t)ldst,   // generic LDS ptr low 32b = LDS offset
        0, 0);
#else
    *(bf16x8*)ldst = *(const bf16x8*)gsrc;
#endif
}

#if HAVE_ASYNC
# if __has_builtin(__builtin_amdgcn_s_wait_asynccnt)
#  define WAIT_ASYNC(n) __builtin_amdgcn_s_wait_asynccnt(n)
# else
#  define WAIT_ASYNC(n) asm volatile("s_wait_asynccnt %0" :: "i"(n))
# endif
#else
# define WAIT_ASYNC(n) ((void)0)
#endif

// ---------------------------------------------------------------------------
// WMMA helpers (CDNA5 layouts, wave32)
// ---------------------------------------------------------------------------
__device__ __forceinline__ floatx8 wmma_bf16(bf16x16 a, bf16x16 b, floatx8 c) {
    return __builtin_amdgcn_wmma_f32_16x16x32_bf16(
        false, a, false, b, (short)0, c, false, false);
}

// A fragment 16x32 from row-major [row][k], row stride ld (bf16 elems).
__device__ __forceinline__ bf16x16 load_a_frag(const bf16* base, int M0, int K0,
                                               int ld, int lane) {
    int row  = M0 + (lane & 15);
    int half = lane >> 4;
    const bf16* p = base + row * ld + K0 + half * 8;
    bf16x8 lo = *(const bf16x8*)p;         // K = K0+8h .. +7
    bf16x8 hi = *(const bf16x8*)(p + 16);  // K = K0+16+8h .. +7
    return __builtin_shufflevector(lo, hi, 0,1,2,3,4,5,6,7,8,9,10,11,12,13,14,15);
}

// B fragment 32x16 from n-major [n][k], row stride ld (works for LDS or global).
__device__ __forceinline__ bf16x16 load_b_frag(const bf16* base, int N0, int K0,
                                               int ld, int lane) {
    int row  = N0 + (lane & 15);
    int half = lane >> 4;
    const bf16* p = base + row * ld + K0 + half * 8;
    bf16x8 lo = *(const bf16x8*)p;
    bf16x8 hi = *(const bf16x8*)(p + 16);
    return __builtin_shufflevector(lo, hi, 0,1,2,3,4,5,6,7,8,9,10,11,12,13,14,15);
}

__device__ __forceinline__ void zero_acc(floatx8 acc[8]) {
#pragma unroll
    for (int i = 0; i < 8; ++i)
#pragma unroll
        for (int j = 0; j < 8; ++j) acc[i][j] = 0.0f;
}

// D = Xs(64x256, LDS) @ W^T(256 o-rows x 256 c, n-major in L2) ; wave (wm,wn)
// accumulates its 16x128 slice.
__device__ __forceinline__ void gemm_tile_accum(floatx8 acc[8], const bf16* Xs,
                                                const bf16* W, int wm, int wn,
                                                int lane) {
#pragma unroll
    for (int k0 = 0; k0 < 256; k0 += 32) {
        bf16x16 a = load_a_frag(Xs, wm * 16, k0, QLD, lane);
#pragma unroll
        for (int ct = 0; ct < 8; ++ct) {
            bf16x16 bb = load_b_frag(W, wn * 128 + ct * 16, k0, C_, lane);
            acc[ct] = wmma_bf16(a, bb, acc[ct]);
        }
    }
}

// ---------------------------------------------------------------------------
// 0) small converters / transposes
// ---------------------------------------------------------------------------
__global__ void zero_stats_kernel(float* stats) { stats[threadIdx.x] = 0.0f; }

// convert the four 256x256 weight matrices to bf16
__global__ __launch_bounds__(256)
void convw_kernel(const float* __restrict__ a0, const float* __restrict__ a1,
                  const float* __restrict__ a2, const float* __restrict__ a3,
                  bf16* __restrict__ o0, bf16* __restrict__ o1,
                  bf16* __restrict__ o2, bf16* __restrict__ o3) {
    int idx = blockIdx.x * 256 + threadIdx.x;      // < 4*65536
    int m = idx >> 16, e = idx & 65535;
    const float* s = (m == 0) ? a0 : (m == 1) ? a1 : (m == 2) ? a2 : a3;
    bf16* d        = (m == 0) ? o0 : (m == 1) ? o1 : (m == 2) ? o2 : o3;
    d[e] = (bf16)s[e];
}

// x (b,c,n) fp32 -> xT (b,n,c) bf16, 64x64 LDS tiles
__global__ __launch_bounds__(256)
void convx_kernel(const float* __restrict__ x, bf16* __restrict__ xT) {
    __shared__ bf16 t[64 * 72];
    const int tid = threadIdx.x;
    const int n0 = blockIdx.x * 64, c0 = blockIdx.y * 64, b = blockIdx.z;
    const float* xb = x + ((size_t)b * C_ + c0) * N_ + n0;
#pragma unroll
    for (int i = 0; i < 16; ++i) {
        int e = tid + i * 256; int nn = e & 63; int cc = e >> 6;
        t[cc * 72 + nn] = (bf16)xb[(size_t)cc * N_ + nn];
    }
    __syncthreads();
    bf16* ob = xT + ((size_t)b * N_ + n0) * C_ + c0;
#pragma unroll
    for (int i = 0; i < 16; ++i) {
        int e = tid + i * 256; int cc = e & 63; int nn = e >> 6;
        ob[(size_t)nn * C_ + cc] = t[cc * 72 + nn];
    }
}

// gN (b,n,ci) -> gT (b,ci,n) bf16, 64x64 LDS tiles
__global__ __launch_bounds__(256)
void transpose_g_kernel(const bf16* __restrict__ gN, bf16* __restrict__ gT) {
    __shared__ bf16 t[64 * 72];
    const int tid = threadIdx.x;
    const int n0 = blockIdx.x * 64, c0 = blockIdx.y * 64, b = blockIdx.z;
    const bf16* src = gN + ((size_t)b * N_ + n0) * CI_ + c0;
#pragma unroll
    for (int i = 0; i < 16; ++i) {
        int e = tid + i * 256; int cc = e & 63; int nn = e >> 6;
        t[nn * 72 + cc] = src[(size_t)nn * CI_ + cc];
    }
    __syncthreads();
    bf16* dst = gT + ((size_t)b * CI_ + c0) * N_ + n0;
#pragma unroll
    for (int i = 0; i < 16; ++i) {
        int e = tid + i * 256; int nn = e & 63; int cc = e >> 6;
        dst[(size_t)cc * N_ + nn] = t[nn * 72 + cc];
    }
}

// ---------------------------------------------------------------------------
// 1) WMMA projections: theta/phi/gN (b,n,ci) bf16 = xT @ W^T + bias
// ---------------------------------------------------------------------------
__global__ __launch_bounds__(256, 1)
void proj_wmma_kernel(const bf16* __restrict__ xT,
                      const bf16* __restrict__ twb, const float* __restrict__ tbias,
                      const bf16* __restrict__ pwb, const float* __restrict__ pbias,
                      const bf16* __restrict__ gwb, const float* __restrict__ gbias,
                      bf16* __restrict__ theta, bf16* __restrict__ phi,
                      bf16* __restrict__ gN) {
    __shared__ __attribute__((aligned(16))) bf16 Xs[BM * QLD];
    const int tid = threadIdx.x, lane = tid & 31, wave = tid >> 5;
    const int wm = wave & 3, wn = wave >> 2;
    const int b = blockIdx.y, n0 = blockIdx.x * BM;
    const bf16* xb = xT + ((size_t)b * N_ + n0) * C_;
#pragma unroll
    for (int i = 0; i < 8; ++i) {
        int idx = tid + i * 256; int r = idx >> 5; int cc = (idx & 31) * 8;
        cp16_async(&Xs[r * QLD + cc], &xb[(size_t)r * C_ + cc]);
    }
    WAIT_ASYNC(0);
    __syncthreads();

    const bf16*  Ws[3] = {twb, pwb, gwb};
    const float* Bs[3] = {tbias, pbias, gbias};
    bf16*        Os[3] = {theta, phi, gN};
#pragma unroll
    for (int w = 0; w < 3; ++w) {
        floatx8 acc[8];
        zero_acc(acc);
        gemm_tile_accum(acc, Xs, Ws[w], wm, wn, lane);
        bf16* out = Os[w] + ((size_t)b * N_ + n0) * CI_;
#pragma unroll
        for (int ct = 0; ct < 8; ++ct) {
            int col = wn * 128 + ct * 16 + (lane & 15);
            float bv = Bs[w][col];
#pragma unroll
            for (int r = 0; r < 8; ++r) {
                int row = wm * 16 + r + ((lane >> 4) << 3);
                out[(size_t)row * CI_ + col] = (bf16)(acc[ct][r] + bv);
            }
        }
    }
}

// ---------------------------------------------------------------------------
// 2) flash attention: y(b,n,ci) bf16 = softmax(theta @ phi^T) @ g
//    async double-buffered K tile, G tile loaded under S-compute/softmax
// ---------------------------------------------------------------------------
__global__ __launch_bounds__(256, 1)
void flash_attn_kernel(const bf16* __restrict__ theta,  // (B,N,CI)
                       const bf16* __restrict__ phi,    // (B,N,CI)
                       const bf16* __restrict__ gT,     // (B,CI,N)
                       bf16* __restrict__ y) {          // (B,N,CI)
    __shared__ __attribute__((aligned(16))) bf16  Qs[BM * QLD];
    __shared__ __attribute__((aligned(16))) bf16  Ks[2][BN * KLD];
    __shared__ __attribute__((aligned(16))) bf16  Gs[CI_ * GLD];
    __shared__ __attribute__((aligned(16))) float SPs[BM * SLD];  // S fp32 / P bf16 alias
    __shared__ float mstate[BM];
    __shared__ float lstate[BM];
    __shared__ float scales[BM];
    bf16* Ps = (bf16*)SPs;   // row r of P starts exactly at row r of S (272B rows)

    const int tid  = threadIdx.x;
    const int lane = tid & 31;
    const int wave = tid >> 5;
    const int wm   = wave & 3;
    const int wn   = wave >> 2;
    const int b    = blockIdx.y;
    const int m0   = blockIdx.x * BM;

    const bf16* thetab = theta + (size_t)b * N_ * CI_;
    const bf16* phib   = phi   + (size_t)b * N_ * CI_;
    const bf16* gbp    = gT    + (size_t)b * CI_ * N_;

    // prologue: async load Q tile and K tile 0
#pragma unroll
    for (int i = 0; i < 8; ++i) {
        int idx = tid + i * 256; int r = idx >> 5; int cc = (idx & 31) * 8;
        cp16_async(&Qs[r * QLD + cc], &thetab[(size_t)(m0 + r) * CI_ + cc]);
    }
#pragma unroll
    for (int i = 0; i < 8; ++i) {
        int idx = tid + i * 256; int r = idx >> 5; int cc = (idx & 31) * 8;
        cp16_async(&Ks[0][r * KLD + cc], &phib[(size_t)r * CI_ + cc]);
    }
    if (tid < BM) { mstate[tid] = -3.0e38f; lstate[tid] = 0.0f; }

    floatx8 oacc[8];
    zero_acc(oacc);

    WAIT_ASYNC(0);
    __syncthreads();

    const int NT = N_ / BN;
    for (int kt = 0; kt < NT; ++kt) {
        const int cur = kt & 1, nxt = cur ^ 1;
        const int key0 = kt * BN;

        // async: G tile for THIS iteration (consumed after softmax)
#pragma unroll
        for (int i = 0; i < 8; ++i) {
            int idx = tid + i * 256; int cch = idx >> 3; int mm = (idx & 7) * 8;
            cp16_async(&Gs[cch * GLD + mm], &gbp[(size_t)cch * N_ + key0 + mm]);
        }
        // async: prefetch NEXT K tile into the other buffer
        if (kt + 1 < NT) {
            const int k2 = (kt + 1) * BN;
#pragma unroll
            for (int i = 0; i < 8; ++i) {
                int idx = tid + i * 256; int r = idx >> 5; int cc = (idx & 31) * 8;
                cp16_async(&Ks[nxt][r * KLD + cc],
                           &phib[(size_t)(k2 + r) * CI_ + cc]);
            }
        }

        // --- S = Q @ K^T from Ks[cur] (overlaps the async DMAs above)
        floatx8 sacc[2];
#pragma unroll
        for (int t = 0; t < 2; ++t)
#pragma unroll
            for (int j = 0; j < 8; ++j) sacc[t][j] = 0.0f;
#pragma unroll
        for (int k0 = 0; k0 < D_; k0 += 32) {
            bf16x16 a = load_a_frag(Qs, wm * 16, k0, QLD, lane);
#pragma unroll
            for (int t = 0; t < 2; ++t) {
                bf16x16 bb = load_b_frag(Ks[cur], wn * 32 + t * 16, k0, KLD, lane);
                sacc[t] = wmma_bf16(a, bb, sacc[t]);
            }
        }
#pragma unroll
        for (int t = 0; t < 2; ++t)
#pragma unroll
            for (int r = 0; r < 8; ++r) {
                int row = wm * 16 + r + ((lane >> 4) << 3);
                int col = wn * 32 + t * 16 + (lane & 15);
                SPs[row * SLD + col] = sacc[t][r];
            }
        __syncthreads();

        // --- online softmax, one thread per row; P aliases S row-for-row
        if (tid < BM) {
            const float* srow = &SPs[tid * SLD];
            bf16* prow = &Ps[tid * PLD];
            float mprev = mstate[tid], lprev = lstate[tid];
            float mt = mprev;
#pragma unroll
            for (int j = 0; j < BN; j += 4) {
                float4 v = *(const float4*)&srow[j];
                mt = fmaxf(mt, fmaxf(fmaxf(v.x, v.y), fmaxf(v.z, v.w)));
            }
            float corr = __expf(mprev - mt);
            float psum = 0.0f;
#pragma unroll 8
            for (int j = 0; j < BN; ++j) {          // write 2B@2j after read 4B@4j: safe
                float p = __expf(srow[j] - mt);
                psum += p;
                prow[j] = (bf16)p;
            }
            mstate[tid] = mt;
            lstate[tid] = lprev * corr + psum;
            scales[tid] = corr;
        }

        // G must be resident; K prefetch (last 8 issued) may still be in flight
        if (kt + 1 < NT) { WAIT_ASYNC(8); } else { WAIT_ASYNC(0); }
        __syncthreads();

        // --- rescale O, then O += P @ G
#pragma unroll
        for (int r = 0; r < 8; ++r) {
            float f = scales[wm * 16 + r + ((lane >> 4) << 3)];
#pragma unroll
            for (int ct = 0; ct < 8; ++ct) oacc[ct][r] *= f;
        }
#pragma unroll
        for (int k0 = 0; k0 < BN; k0 += 32) {
            bf16x16 a = load_a_frag(Ps, wm * 16, k0, PLD, lane);
#pragma unroll
            for (int ct = 0; ct < 8; ++ct) {
                bf16x16 bb = load_b_frag(Gs, wn * 128 + ct * 16, k0, GLD, lane);
                oacc[ct] = wmma_bf16(a, bb, oacc[ct]);
            }
        }

        WAIT_ASYNC(0);      // next K tile landed
        __syncthreads();    // Gs / S / P consumed by everyone
    }

    // --- final 1/l normalization, store y bf16 (b,n,ci)
    bf16* yb = y + (size_t)b * N_ * CI_;
#pragma unroll
    for (int r = 0; r < 8; ++r) {
        int row   = wm * 16 + r + ((lane >> 4) << 3);
        float inv = 1.0f / lstate[row];
#pragma unroll
        for (int ct = 0; ct < 8; ++ct) {
            int col = wn * 128 + ct * 16 + (lane & 15);
            yb[(size_t)(m0 + row) * CI_ + col] = (bf16)(oacc[ct][r] * inv);
        }
    }
}

// ---------------------------------------------------------------------------
// 3) W conv on WMMA: z (b,n,o) fp32 = y @ w_w^T + w_b
// ---------------------------------------------------------------------------
__global__ __launch_bounds__(256, 1)
void zgemm_wmma_kernel(const bf16* __restrict__ y, const bf16* __restrict__ wwb,
                       const float* __restrict__ wbias, float* __restrict__ z) {
    __shared__ __attribute__((aligned(16))) bf16 Ys[BM * QLD];
    const int tid = threadIdx.x, lane = tid & 31, wave = tid >> 5;
    const int wm = wave & 3, wn = wave >> 2;
    const int b = blockIdx.y, n0 = blockIdx.x * BM;
    const bf16* yb = y + ((size_t)b * N_ + n0) * CI_;
#pragma unroll
    for (int i = 0; i < 8; ++i) {
        int idx = tid + i * 256; int r = idx >> 5; int cc = (idx & 31) * 8;
        cp16_async(&Ys[r * QLD + cc], &yb[(size_t)r * CI_ + cc]);
    }
    WAIT_ASYNC(0);
    __syncthreads();

    floatx8 acc[8];
    zero_acc(acc);
    gemm_tile_accum(acc, Ys, wwb, wm, wn, lane);

    float* out = z + ((size_t)b * N_ + n0) * CI_;
#pragma unroll
    for (int ct = 0; ct < 8; ++ct) {
        int col = wn * 128 + ct * 16 + (lane & 15);
        float bv = wbias[col];
#pragma unroll
        for (int r = 0; r < 8; ++r) {
            int row = wm * 16 + r + ((lane >> 4) << 3);
            out[(size_t)row * CI_ + col] = acc[ct][r] + bv;
        }
    }
}

// ---------------------------------------------------------------------------
// 4) per-channel sums for BatchNorm (coalesced; channel = threadIdx.x)
// ---------------------------------------------------------------------------
__global__ __launch_bounds__(256)
void stats_kernel(const float* __restrict__ z, float* __restrict__ stats) {
    const int o  = threadIdx.x;
    const int r0 = blockIdx.x * 128;           // rows over B*N = 16384
    float s = 0.0f, s2 = 0.0f;
    for (int r = 0; r < 128; ++r) {
        float v = z[(size_t)(r0 + r) * CI_ + o];
        s += v; s2 += v * v;
    }
    atomicAdd(&stats[o], s);
    atomicAdd(&stats[CI_ + o], s2);
}

// ---------------------------------------------------------------------------
// 5) BatchNorm + transpose to output layout (b,ci,n), 64x64 LDS tiles
// ---------------------------------------------------------------------------
__global__ __launch_bounds__(256)
void bn_t_kernel(const float* __restrict__ z, const float* __restrict__ stats,
                 const float* __restrict__ gamma, const float* __restrict__ beta,
                 float* __restrict__ out) {
    __shared__ float t[64 * 68];
    const int tid = threadIdx.x;
    const int n0 = blockIdx.x * 64, o0 = blockIdx.y * 64, b = blockIdx.z;
    const float cnt = (float)(B_ * N_);
#pragma unroll
    for (int i = 0; i < 16; ++i) {
        int e = tid + i * 256; int oo = e & 63; int nn = e >> 6;
        int o = o0 + oo;
        float mean = stats[o] / cnt;
        float var  = stats[CI_ + o] / cnt - mean * mean;
        float v = z[((size_t)b * N_ + n0 + nn) * CI_ + o];
        t[nn * 68 + oo] = gamma[o] * (v - mean) * rsqrtf(var + EPS_) + beta[o];
    }
    __syncthreads();
#pragma unroll
    for (int i = 0; i < 16; ++i) {
        int e = tid + i * 256; int nn = e & 63; int oo = e >> 6;
        out[((size_t)b * CI_ + o0 + oo) * N_ + n0 + nn] = t[nn * 68 + oo];
    }
}

// ---------------------------------------------------------------------------
// host launcher
// ---------------------------------------------------------------------------
extern "C" void kernel_launch(void* const* d_in, const int* in_sizes, int n_in,
                              void* d_out, int out_size, void* d_ws, size_t ws_size,
                              hipStream_t stream) {
    const float* x    = (const float*)d_in[0];
    const float* g_w  = (const float*)d_in[1];
    const float* g_b  = (const float*)d_in[2];
    const float* th_w = (const float*)d_in[3];
    const float* th_b = (const float*)d_in[4];
    const float* ph_w = (const float*)d_in[5];
    const float* ph_b = (const float*)d_in[6];
    const float* w_w  = (const float*)d_in[7];
    const float* w_b  = (const float*)d_in[8];
    const float* bng  = (const float*)d_in[9];
    const float* bnb  = (const float*)d_in[10];
    float* out = (float*)d_out;

    char* ws = (char*)d_ws;
    size_t off = 0;
    auto take = [&](size_t bytes) {
        char* p = ws + off;
        off = (off + bytes + 255) & ~(size_t)255;
        return p;
    };
    bf16*  twb   = (bf16*)take(65536 * sizeof(bf16));
    bf16*  pwb   = (bf16*)take(65536 * sizeof(bf16));
    bf16*  gwb   = (bf16*)take(65536 * sizeof(bf16));
    bf16*  wwb   = (bf16*)take(65536 * sizeof(bf16));
    bf16*  xT    = (bf16*)take((size_t)B_ * N_ * C_  * sizeof(bf16));
    bf16*  theta = (bf16*)take((size_t)B_ * N_ * CI_ * sizeof(bf16));
    bf16*  phi   = (bf16*)take((size_t)B_ * N_ * CI_ * sizeof(bf16));
    bf16*  gN    = (bf16*)take((size_t)B_ * N_ * CI_ * sizeof(bf16));
    bf16*  gT    = (bf16*)take((size_t)B_ * CI_ * N_ * sizeof(bf16));
    bf16*  y     = (bf16*)take((size_t)B_ * N_ * CI_ * sizeof(bf16));
    float* z     = (float*)take((size_t)B_ * N_ * CI_ * sizeof(float));
    float* stats = (float*)take(2 * CI_ * sizeof(float));

    convw_kernel<<<1024, 256, 0, stream>>>(th_w, ph_w, g_w, w_w, twb, pwb, gwb, wwb);
    convx_kernel<<<dim3(N_ / 64, C_ / 64, B_), 256, 0, stream>>>(x, xT);
    proj_wmma_kernel<<<dim3(N_ / BM, B_), 256, 0, stream>>>(
        xT, twb, th_b, pwb, ph_b, gwb, g_b, theta, phi, gN);
    transpose_g_kernel<<<dim3(N_ / 64, CI_ / 64, B_), 256, 0, stream>>>(gN, gT);
    flash_attn_kernel<<<dim3(N_ / BM, B_), 256, 0, stream>>>(theta, phi, gT, y);
    zgemm_wmma_kernel<<<dim3(N_ / BM, B_), 256, 0, stream>>>(y, wwb, w_b, z);
    zero_stats_kernel<<<1, 2 * CI_, 0, stream>>>(stats);
    stats_kernel<<<(B_ * N_) / 128, 256, 0, stream>>>(z, stats);
    bn_t_kernel<<<dim3(N_ / 64, CI_ / 64, B_), 256, 0, stream>>>(z, stats, bng, bnb, out);
}
